// ROPEAttnHead_2671469658750
// MI455X (gfx1250) — compile-verified, matrix operand order checked by reading, not run on hardware
//
#include <hip/hip_runtime.h>
#include <hip/hip_bf16.h>

typedef __attribute__((ext_vector_type(16))) __bf16 v16bf;
typedef __attribute__((ext_vector_type(8)))  __bf16 v8bf;
typedef __attribute__((ext_vector_type(4)))  __bf16 v4bf;
typedef __attribute__((ext_vector_type(2)))  __bf16 v2bf;
typedef __attribute__((ext_vector_type(8)))  float  v8f;
typedef __attribute__((ext_vector_type(4)))  int    v4i;

#define EMB    2048
#define HD     128
#define SEQN   4096
#define NBATCH 4
#define QSCALE 0.08838834764831845f  // 1/sqrt(128), folded into Q

#define AS1 __attribute__((address_space(1)))
#define AS3 __attribute__((address_space(3)))

#if __has_builtin(__builtin_amdgcn_global_load_async_to_lds_b128)
#define HAVE_ASYNC_LDS 1
#else
#define HAVE_ASYNC_LDS 0
#endif

__device__ __forceinline__ void wait_async_all() {
#if HAVE_ASYNC_LDS
#if __has_builtin(__builtin_amdgcn_s_wait_asynccnt)
    __builtin_amdgcn_s_wait_asynccnt(0);
#else
    asm volatile("s_wait_asynccnt 0" ::: "memory");
#endif
#endif
}

// Build a 16-element bf16 fragment from two contiguous 8-element (16B) runs.
// Lowered as two 128-bit loads (ds_load_b128 / global_load_b128).
__device__ __forceinline__ v16bf make_frag(const __bf16* lo, const __bf16* hi) {
    v8bf l = *(const v8bf*)lo;
    v8bf h = *(const v8bf*)hi;
    return __builtin_shufflevector(l, h, 0, 1, 2, 3, 4, 5, 6, 7,
                                         8, 9, 10, 11, 12, 13, 14, 15);
}

// ---------------------------------------------------------------------------
// Kernel 1: QKV projection (bf16 WMMA, f32 accum) + bias + RoPE, writes bf16.
// grid = (B*S/64, 3), block = 256 (8 waves). Wave w owns N cols [16w,16w+16),
// iterates 4 M-subtiles (64 rows/block) sharing one B fragment. X/W tiles are
// double-buffered: stage k-tile t+1 while WMMAs run on tile t (1 barrier/iter).
// ---------------------------------------------------------------------------
__global__ __launch_bounds__(256)
void qkv_rope_kernel(const float* __restrict__ x,
                     const float* __restrict__ Wq, const float* __restrict__ bq,
                     const float* __restrict__ Wk, const float* __restrict__ bk,
                     const float* __restrict__ Wv, const float* __restrict__ bv,
                     __bf16* __restrict__ Qb, __bf16* __restrict__ Kb,
                     __bf16* __restrict__ Vb)
{
    const int mat     = blockIdx.y;          // 0=Q, 1=K, 2=V
    const int rowBase = blockIdx.x * 64;     // over B*S flattened rows
    const float* W    = (mat == 0) ? Wq : ((mat == 1) ? Wk : Wv);
    const float* bias = (mat == 0) ? bq : ((mat == 1) ? bk : bv);
    __bf16*      outp = (mat == 0) ? Qb : ((mat == 1) ? Kb : Vb);

    __shared__ __align__(16) __bf16 ldsX[2][64][32];   // A tiles (k fast)
    __shared__ __align__(16) __bf16 ldsWt[2][HD][32];  // B tiles, transposed
    __shared__ float ldsC[64][HD];

    const int tid  = threadIdx.x;
    const int wave = tid >> 5;
    const int lane = tid & 31;
    const int rowA = lane & 15;
    const int hi8  = (lane >> 4) * 8;
    const int nCol = wave * 16 + rowA;

    auto stage = [&](int k0, int buf) {
        // X tile 64x32 (fp32 -> bf16): 2x float4 loads per thread
#pragma unroll
        for (int it = 0; it < 2; ++it) {
            const int i = tid + it * 256;
            const int r = i >> 3, c4 = (i & 7) * 4;
            const float4 xv = *(const float4*)&x[(size_t)(rowBase + r) * EMB + k0 + c4];
            v4bf b4 = { (__bf16)xv.x, (__bf16)xv.y, (__bf16)xv.z, (__bf16)xv.w };
            *(v4bf*)&ldsX[buf][r][c4] = b4;
        }
        // W tile 32x128 transposed -> ldsWt[n][k], k-pairs as 4B stores
#pragma unroll
        for (int it = 0; it < 2; ++it) {
            const int i = tid + it * 256;
            const int rp = i >> 5, c4 = (i & 31) * 4;
            const float4 wa = *(const float4*)&W[(size_t)(k0 + 2 * rp) * HD + c4];
            const float4 wb = *(const float4*)&W[(size_t)(k0 + 2 * rp + 1) * HD + c4];
            v2bf p0 = { (__bf16)wa.x, (__bf16)wb.x };
            v2bf p1 = { (__bf16)wa.y, (__bf16)wb.y };
            v2bf p2 = { (__bf16)wa.z, (__bf16)wb.z };
            v2bf p3 = { (__bf16)wa.w, (__bf16)wb.w };
            *(v2bf*)&ldsWt[buf][c4 + 0][2 * rp] = p0;
            *(v2bf*)&ldsWt[buf][c4 + 1][2 * rp] = p1;
            *(v2bf*)&ldsWt[buf][c4 + 2][2 * rp] = p2;
            *(v2bf*)&ldsWt[buf][c4 + 3][2 * rp] = p3;
        }
    };

    v8f acc[4];
    const v8f vzero = {};
#pragma unroll
    for (int mt = 0; mt < 4; ++mt) acc[mt] = vzero;

    stage(0, 0);
    __syncthreads();

    const int NT = EMB / 32;   // 64 k-tiles
    for (int kt = 0; kt < NT; ++kt) {
        const int cur = kt & 1;
        if (kt + 1 < NT) stage((kt + 1) * 32, cur ^ 1);

        v16bf bfr = make_frag(&ldsWt[cur][nCol][hi8], &ldsWt[cur][nCol][16 + hi8]);
#pragma unroll
        for (int mt = 0; mt < 4; ++mt) {
            v16bf a = make_frag(&ldsX[cur][mt * 16 + rowA][hi8],
                                &ldsX[cur][mt * 16 + rowA][16 + hi8]);
            acc[mt] = __builtin_amdgcn_wmma_f32_16x16x32_bf16(false, a, false, bfr,
                                                              (short)0, acc[mt],
                                                              false, false);
        }
        __syncthreads();   // cur consumed by all; nxt staged and visible
    }

    // C fragments -> LDS (rows mt*16 + r + 8*hi, col = nCol)
#pragma unroll
    for (int mt = 0; mt < 4; ++mt)
#pragma unroll
        for (int r = 0; r < 8; ++r)
            ldsC[mt * 16 + r + hi8][nCol] = acc[mt][r];
    __syncthreads();

    // bias + RoPE (Q,K) + bf16 store (4B pair stores); fold 1/sqrt(D) into Q
    const float LN1E4 = 9.210340371976184f;
#pragma unroll 4
    for (int it = 0; it < 16; ++it) {
        const int i  = tid + it * 256;
        const int r  = i >> 6;
        const int pr = i & 63;
        const int gRow = rowBase + r;
        const int s    = gRow & (SEQN - 1);
        float x0 = ldsC[r][2 * pr]     + bias[2 * pr];
        float x1 = ldsC[r][2 * pr + 1] + bias[2 * pr + 1];
        float o0, o1;
        if (mat < 2) {
            float invf = __expf(-((float)(2 * pr) / (float)HD) * LN1E4);
            float ang  = (float)s * invf;
            float c = cosf(ang), sn = sinf(ang);
            o0 = x0 * c - x1 * sn;
            o1 = x1 * c + x0 * sn;
            if (mat == 0) { o0 *= QSCALE; o1 *= QSCALE; }
        } else {
            o0 = x0; o1 = x1;
        }
        v2bf pk = { (__bf16)o0, (__bf16)o1 };
        *(v2bf*)&outp[(size_t)gRow * HD + 2 * pr] = pk;
    }
}

// ---------------------------------------------------------------------------
// Kernel 2: causal flash attention. grid = B*(S/128), block = 256 (8 waves).
// Wave w owns q rows [qBlock+16w, +16). 32-key KV tiles double-buffered in
// LDS: K staged via async global->LDS b128 (ASYNCcnt ping-pong), V staged
// transposed (k fast). Tile t+1 traffic overlaps tile t's 16 WMMAs.
// ---------------------------------------------------------------------------
__global__ __launch_bounds__(256)
void flash_attn_kernel(const __bf16* __restrict__ Qb,
                       const __bf16* __restrict__ Kb,
                       const __bf16* __restrict__ Vb,
                       float* __restrict__ out)
{
    const int b      = blockIdx.x / (SEQN / 128);
    const int qBlock = (blockIdx.x % (SEQN / 128)) * 128;
    const int tid  = threadIdx.x;
    const int wave = tid >> 5;
    const int lane = tid & 31;
    const int rowA = lane & 15;
    const int hi8  = (lane >> 4) * 8;
    const int qBase = qBlock + wave * 16;

    const __bf16* Q = Qb + (size_t)b * SEQN * HD;
    const __bf16* K = Kb + (size_t)b * SEQN * HD;
    const __bf16* V = Vb + (size_t)b * SEQN * HD;

    __shared__ __align__(16) __bf16 ldsK[2][32][HD];   // row-major: d fast
    __shared__ __align__(16) __bf16 ldsVt[2][HD][32];  // transposed: k fast
    __shared__ __align__(16) __bf16 ldsP[8][16][32];

    auto stageK = [&](int t, int buf) {
#pragma unroll
        for (int it = 0; it < 2; ++it) {
            const int i = tid + it * 256;
            const int r = i >> 4, c8 = (i & 15) * 8;
#if HAVE_ASYNC_LDS
            __builtin_amdgcn_global_load_async_to_lds_b128(
                (AS1 v4i*)&K[(size_t)(t * 32 + r) * HD + c8],
                (AS3 v4i*)&ldsK[buf][r][c8], 0, 0);
#else
            *(v8bf*)&ldsK[buf][r][c8] =
                *(const v8bf*)&K[(size_t)(t * 32 + r) * HD + c8];
#endif
        }
    };
    auto stageV = [&](int t, int buf) {
        const int rp = tid >> 4, c8 = (tid & 15) * 8;
        const v8bf va = *(const v8bf*)&V[(size_t)(t * 32 + 2 * rp) * HD + c8];
        const v8bf vb = *(const v8bf*)&V[(size_t)(t * 32 + 2 * rp + 1) * HD + c8];
#pragma unroll
        for (int j = 0; j < 8; ++j) {
            v2bf p = { va[j], vb[j] };
            *(v2bf*)&ldsVt[buf][c8 + j][2 * rp] = p;
        }
    };

    // Load Q fragments once (4 A-frags covering d=0..127), already scaled.
    const __bf16* qRow = Q + (size_t)(qBase + rowA) * HD;
    v16bf Aq[4];
#pragma unroll
    for (int f = 0; f < 4; ++f)
        Aq[f] = make_frag(qRow + f * 32 + hi8, qRow + f * 32 + 16 + hi8);

    v8f acc[8];
    const v8f vzero = {};
#pragma unroll
    for (int i = 0; i < 8; ++i) acc[i] = vzero;
    float m[8], l[8];
#pragma unroll
    for (int r = 0; r < 8; ++r) { m[r] = -3.0e38f; l[r] = 0.f; }

    const int T = (qBlock + 128) / 32;   // causal tile count, uniform per block

    stageK(0, 0);
    stageV(0, 0);
    wait_async_all();
    __syncthreads();

    for (int t = 0; t < T; ++t) {
        const int cur = t & 1;
        const int k0  = t * 32;

        // stage tile t+1 into the other buffer (last read at iteration t-1,
        // separated from these writes by the end-of-(t-1) barrier)
        if (t + 1 < T) {
            stageK(t + 1, cur ^ 1);
            stageV(t + 1, cur ^ 1);
        }

        // scores S = Q K^T for two 16-key subtiles
        v8f sfrag[2];
#pragma unroll
        for (int sub = 0; sub < 2; ++sub) {
            v8f sacc = vzero;
            const __bf16* kRow = &ldsK[cur][sub * 16 + rowA][0];
#pragma unroll
            for (int f = 0; f < 4; ++f) {
                v16bf bK = make_frag(kRow + f * 32 + hi8, kRow + f * 32 + 16 + hi8);
                sacc = __builtin_amdgcn_wmma_f32_16x16x32_bf16(false, Aq[f], false, bK,
                                                               (short)0, sacc, false, false);
            }
            sfrag[sub] = sacc;
        }

        // causal mask + online softmax (row state replicated in 16-lane group)
        const int kp0 = k0 + rowA;
        const int kp1 = k0 + 16 + rowA;
        float rowmax[8];
#pragma unroll
        for (int r = 0; r < 8; ++r) {
            const int q = qBase + r + hi8;
            if (kp0 > q) sfrag[0][r] = -3.0e38f;
            if (kp1 > q) sfrag[1][r] = -3.0e38f;
            rowmax[r] = fmaxf(sfrag[0][r], sfrag[1][r]);
        }
#pragma unroll
        for (int d = 1; d < 16; d <<= 1)
#pragma unroll
            for (int r = 0; r < 8; ++r)
                rowmax[r] = fmaxf(rowmax[r], __shfl_xor(rowmax[r], d, 32));

        float rowsum[8];
#pragma unroll
        for (int r = 0; r < 8; ++r) {
            const int q = qBase + r + hi8;
            float mnew = fmaxf(m[r], rowmax[r]);
            float sc   = __expf(m[r] - mnew);
            float p0 = (kp0 > q) ? 0.f : __expf(sfrag[0][r] - mnew);
            float p1 = (kp1 > q) ? 0.f : __expf(sfrag[1][r] - mnew);
            sfrag[0][r] = p0; sfrag[1][r] = p1;
            rowsum[r] = p0 + p1;
            m[r] = mnew;
            l[r] *= sc;
#pragma unroll
            for (int ds = 0; ds < 8; ++ds) acc[ds][r] *= sc;
        }
#pragma unroll
        for (int d = 1; d < 16; d <<= 1)
#pragma unroll
            for (int r = 0; r < 8; ++r)
                rowsum[r] += __shfl_xor(rowsum[r], d, 32);
#pragma unroll
        for (int r = 0; r < 8; ++r) l[r] += rowsum[r];

        // transpose P (C-layout -> A-layout) through per-wave LDS scratch
#pragma unroll
        for (int r = 0; r < 8; ++r) {
            ldsP[wave][r + hi8][rowA]      = (__bf16)sfrag[0][r];
            ldsP[wave][r + hi8][16 + rowA] = (__bf16)sfrag[1][r];
        }
        __syncthreads();

        v16bf Ap = make_frag(&ldsP[wave][rowA][hi8], &ldsP[wave][rowA][16 + hi8]);
#pragma unroll
        for (int ds = 0; ds < 8; ++ds) {
            const __bf16* vRow = &ldsVt[cur][ds * 16 + rowA][0];
            v16bf bV = make_frag(vRow + hi8, vRow + 16 + hi8);
            acc[ds] = __builtin_amdgcn_wmma_f32_16x16x32_bf16(false, Ap, false, bV,
                                                              (short)0, acc[ds], false, false);
        }

        // tile t+1 async K landed; staging visible; all waves done with cur
        wait_async_all();
        __syncthreads();
    }

    // epilogue: normalize and store fp32 [B,S,D]
#pragma unroll
    for (int r = 0; r < 8; ++r) {
        const int q = qBase + r + hi8;
        const float inv = 1.0f / l[r];
#pragma unroll
        for (int ds = 0; ds < 8; ++ds) {
            const int c = ds * 16 + rowA;
            out[((size_t)b * SEQN + q) * HD + c] = acc[ds][r] * inv;
        }
    }
}

// ---------------------------------------------------------------------------
extern "C" void kernel_launch(void* const* d_in, const int* in_sizes, int n_in,
                              void* d_out, int out_size, void* d_ws, size_t ws_size,
                              hipStream_t stream) {
    const float* x  = (const float*)d_in[0];
    const float* Wq = (const float*)d_in[1];
    const float* bq = (const float*)d_in[2];
    const float* Wk = (const float*)d_in[3];
    const float* bk = (const float*)d_in[4];
    const float* Wv = (const float*)d_in[5];
    const float* bv = (const float*)d_in[6];
    float* out = (float*)d_out;

    const size_t n = (size_t)NBATCH * SEQN * HD;       // 2 Mi elements
    char* ws = (char*)d_ws;
    __bf16* Qb = (__bf16*)(ws);
    __bf16* Kb = (__bf16*)(ws + n * sizeof(__bf16));
    __bf16* Vb = (__bf16*)(ws + 2 * n * sizeof(__bf16));

    qkv_rope_kernel<<<dim3((NBATCH * SEQN) / 64, 3), 256, 0, stream>>>(
        x, Wq, bq, Wk, bk, Wv, bv, Qb, Kb, Vb);
    flash_attn_kernel<<<dim3(NBATCH * (SEQN / 128)), 256, 0, stream>>>(
        Qb, Kb, Vb, out);
}
